// VisionTransformer_50723563766248
// MI455X (gfx1250) — matrix operand order repaired and checked by reading, not compiled, not run on hardware
//
#include <hip/hip_runtime.h>

// ---------------------------------------------------------------------------
// ViT-Base forward for MI455X (gfx1250, wave32, WMMA).
// B=4, H=W=512, P=16, E=768, DEPTH=12, NH=12, HEAD=64, HID=3072, N=1025 tok.
// Token rows flattened across batch: 4*1025 = 4100, padded to 4224 (33*128).
// GEMMs: f16 WMMA f32-acc, WG tile 128x128, double-buffered LDS, async
// A-tile copies (ASYNCcnt), transposed B staging -> all frag reads b128.
// Attention: flash-style, one wave per (b, head, 16-query tile).
// ---------------------------------------------------------------------------

typedef __attribute__((ext_vector_type(16))) _Float16 v16h;
typedef __attribute__((ext_vector_type(8)))  float    v8f;
typedef __attribute__((ext_vector_type(4)))  unsigned int v4u;

union F8  { v4u v; _Float16 h[8]; };
union V16 { v16h h; v4u u[2]; };
union PK2 { _Float16 h[2]; unsigned u; };

#define NTOK   1025
#define MROWS  4100   // 4 * 1025 valid token rows
#define MPAD   4224   // 33 * 128 padded rows
#define DEPTHc 12
#define NHc    12
#define HIDc   3072
#define BT_LD  40     // transposed-B LDS row stride (f16): 16B-aligned, no bank clash

// ---------------- elementwise helpers ----------------

__global__ void cvt_f32_to_f16(const float* __restrict__ src,
                               _Float16* __restrict__ dst, long n) {
  long i = (long)blockIdx.x * blockDim.x + threadIdx.x;
  if (i < n) dst[i] = (_Float16)src[i];
}

__global__ void cvt_transpose_768(const float* __restrict__ w,
                                  _Float16* __restrict__ wt) {
  long i = (long)blockIdx.x * blockDim.x + threadIdx.x;
  if (i >= (long)768 * 768) return;
  int k = (int)(i / 768), nn = (int)(i % 768);
  wt[i] = (_Float16)w[(long)nn * 768 + k];
}

__global__ void im2col_kernel(const float* __restrict__ inp,
                              _Float16* __restrict__ pm) {
  long i = (long)blockIdx.x * blockDim.x + threadIdx.x;
  if (i >= (long)MPAD * 768) return;
  int row = (int)(i / 768), k = (int)(i % 768);
  if (row >= MROWS) { pm[i] = (_Float16)0.f; return; }
  int b = row / NTOK, t = row % NTOK;
  if (t == 0) { pm[i] = (_Float16)0.f; return; }
  int p = t - 1, hp = p >> 5, wp = p & 31;
  int c = k >> 8, py = (k >> 4) & 15, px = k & 15;
  pm[i] = (_Float16)inp[(((long)b * 3 + c) * 512 + hp * 16 + py) * 512 + wp * 16 + px];
}

__global__ void embed_fix_kernel(float* __restrict__ x,
                                 const float* __restrict__ cls,
                                 const float* __restrict__ pos) {
  long i = (long)blockIdx.x * blockDim.x + threadIdx.x;
  if (i >= (long)MPAD * 768) return;
  int row = (int)(i / 768), e = (int)(i % 768);
  if (row >= MROWS) { x[i] = 0.f; return; }
  int t = row % NTOK;
  if (t == 0) x[i] = cls[e] + pos[e];
  else        x[i] += pos[(long)t * 768 + e];
}

// ---------------- LayerNorm -> f16 activation ----------------

__global__ __launch_bounds__(256)
void ln_kernel(const float* __restrict__ x, const float* __restrict__ g,
               const float* __restrict__ bta, _Float16* __restrict__ out) {
  int row = blockIdx.x;
  _Float16* orow = out + (long)row * 768;
  if (row >= MROWS) {
    for (int i = threadIdx.x; i < 768; i += 256) orow[i] = (_Float16)0.f;
    return;
  }
  __shared__ float red[256];
  const float* xr = x + (long)row * 768;
  float s = 0.f;
  for (int i = threadIdx.x; i < 768; i += 256) s += xr[i];
  red[threadIdx.x] = s; __syncthreads();
  for (int o = 128; o > 0; o >>= 1) {
    if (threadIdx.x < o) red[threadIdx.x] += red[threadIdx.x + o];
    __syncthreads();
  }
  float mean = red[0] * (1.f / 768.f);
  __syncthreads();
  float s2 = 0.f;
  for (int i = threadIdx.x; i < 768; i += 256) {
    float d = xr[i] - mean; s2 += d * d;
  }
  red[threadIdx.x] = s2; __syncthreads();
  for (int o = 128; o > 0; o >>= 1) {
    if (threadIdx.x < o) red[threadIdx.x] += red[threadIdx.x + o];
    __syncthreads();
  }
  float rstd = rsqrtf(red[0] * (1.f / 768.f) + 1e-6f);
  for (int i = threadIdx.x; i < 768; i += 256)
    orow[i] = (_Float16)((xr[i] - mean) * rstd * g[i] + bta[i]);
}

__global__ __launch_bounds__(256)
void final_kernel(const float* __restrict__ x, const float* __restrict__ g,
                  const float* __restrict__ bta, float* __restrict__ out) {
  int pid = blockIdx.x;              // b*1024 + hp*32 + wp
  int b = pid >> 10, sp = pid & 1023;
  long row = (long)b * NTOK + 1 + sp;
  __shared__ float red[256];
  const float* xr = x + row * 768;
  float s = 0.f;
  for (int i = threadIdx.x; i < 768; i += 256) s += xr[i];
  red[threadIdx.x] = s; __syncthreads();
  for (int o = 128; o > 0; o >>= 1) {
    if (threadIdx.x < o) red[threadIdx.x] += red[threadIdx.x + o];
    __syncthreads();
  }
  float mean = red[0] * (1.f / 768.f);
  __syncthreads();
  float s2 = 0.f;
  for (int i = threadIdx.x; i < 768; i += 256) {
    float d = xr[i] - mean; s2 += d * d;
  }
  red[threadIdx.x] = s2; __syncthreads();
  for (int o = 128; o > 0; o >>= 1) {
    if (threadIdx.x < o) red[threadIdx.x] += red[threadIdx.x + o];
    __syncthreads();
  }
  float rstd = rsqrtf(red[0] * (1.f / 768.f) + 1e-6f);
  for (int e = threadIdx.x; e < 768; e += 256) {
    float v = (xr[e] - mean) * rstd * g[e] + bta[e];
    out[((long)b * 768 + e) * 1024 + sp] = v;
  }
}

// ---------------- generic WMMA GEMM ----------------
// C[MPAD,N] = A[MPAD,K](f16) @ Bw[K,N](f16) + bias, opt residual / GELU.
// 256 thr (8 waves), WG tile 128(M) x 128(N), k-step 32, double-buffered.
// Wave w: rows (w&3)*32..+31 (2 A-frags), cols (w>>2)*64..+63 (4 B-frags)
// -> 8 WMMAs per k-step with all fragment loads hoisted before the chain.
// flags: 1=GELU, 2=+resid, 4=store f32, 8=store f16.

__global__ __launch_bounds__(256)
void gemm_ws(const _Float16* __restrict__ A, const _Float16* __restrict__ Bw,
             const float* __restrict__ bias, const float* resid,
             float* Cf, _Float16* Ch, int N, int K, int flags) {
  __shared__ alignas(16) _Float16 lA[2][128 * 32];
  __shared__ alignas(16) _Float16 lBt[2][128 * BT_LD];
  const int t = threadIdx.x;
  const int lane = t & 31, w = t >> 5;
  const int mt = w & 3;        // wave's 32-row slab inside WG (0..3)
  const int nh = w >> 2;       // 0/1 -> 64-col half
  const int m0 = blockIdx.y * 128;
  const int n0 = blockIdx.x * 128;
  const int nn = lane & 15, half = lane >> 4;

  // staging coordinates
  const int ar = t >> 1, ac = (t & 1) << 4;       // A: 128 rows x 32 cols, 16 f16/thr
  const int br0 = t >> 4, br1 = 16 + (t >> 4);    // B rows (k) for 2 passes
  const int bc = (t & 15) << 3;                   // B col group (8 f16)
  const _Float16* gA = A + (long)(m0 + ar) * K + ac;

  v8f acc0 = {}, acc1 = {}, acc2 = {}, acc3 = {};
  v8f acc4 = {}, acc5 = {}, acc6 = {}, acc7 = {};

  // ---- prologue: tile 0
  {
    unsigned la = (unsigned)(unsigned long long)(void*)&lA[0][ar * 32 + ac];
    asm volatile("global_load_async_to_lds_b128 %0, %1, off"
                 :: "v"(la), "v"(gA) : "memory");
    asm volatile("global_load_async_to_lds_b128 %0, %1, off"
                 :: "v"(la + 16u), "v"(gA + 8) : "memory");
    F8 f0, f1;
    f0.v = *(const v4u*)(Bw + (long)br0 * N + n0 + bc);
    f1.v = *(const v4u*)(Bw + (long)br1 * N + n0 + bc);
#pragma unroll
    for (int e = 0; e < 8; ++e) {
      lBt[0][(bc + e) * BT_LD + br0] = f0.h[e];
      lBt[0][(bc + e) * BT_LD + br1] = f1.h[e];
    }
    asm volatile("s_wait_asynccnt 0x0" ::: "memory");
  }
  __syncthreads();

  int cur = 0;
  for (int k0 = 0; k0 < K; k0 += 32) {
    const int nxt = cur ^ 1;
    const bool hasNext = (k0 + 32) < K;
    F8 f0, f1;
    if (hasNext) {
      unsigned la = (unsigned)(unsigned long long)(void*)&lA[nxt][ar * 32 + ac];
      const _Float16* ga = gA + (k0 + 32);
      asm volatile("global_load_async_to_lds_b128 %0, %1, off"
                   :: "v"(la), "v"(ga) : "memory");
      asm volatile("global_load_async_to_lds_b128 %0, %1, off"
                   :: "v"(la + 16u), "v"(ga + 8) : "memory");
      f0.v = *(const v4u*)(Bw + (long)(k0 + 32 + br0) * N + n0 + bc);
      f1.v = *(const v4u*)(Bw + (long)(k0 + 32 + br1) * N + n0 + bc);
    }

    // ---- hoist ALL fragment loads, then 8 back-to-back WMMAs
    v16h a0, a1;
    {
      const _Float16* ab0 = &lA[cur][(mt * 32 + nn) * 32];
      const _Float16* ab1 = &lA[cur][(mt * 32 + 16 + nn) * 32];
#pragma unroll
      for (int v = 0; v < 8; ++v) {   // contiguous k = 16*half + s (matches B)
        int k = 2 * v + (half << 4);
        a0[2 * v] = ab0[k]; a0[2 * v + 1] = ab0[k + 1];
        a1[2 * v] = ab1[k]; a1[2 * v + 1] = ab1[k + 1];
      }
    }
    V16 bf0, bf1, bf2, bf3;
    {
      const _Float16* bb = &lBt[cur][(nh * 64 + nn) * BT_LD + 16 * half];
      bf0.u[0] = *(const v4u*)(bb);                  bf0.u[1] = *(const v4u*)(bb + 8);
      bf1.u[0] = *(const v4u*)(bb + 16 * BT_LD);     bf1.u[1] = *(const v4u*)(bb + 16 * BT_LD + 8);
      bf2.u[0] = *(const v4u*)(bb + 32 * BT_LD);     bf2.u[1] = *(const v4u*)(bb + 32 * BT_LD + 8);
      bf3.u[0] = *(const v4u*)(bb + 48 * BT_LD);     bf3.u[1] = *(const v4u*)(bb + 48 * BT_LD + 8);
    }
    acc0 = __builtin_amdgcn_wmma_f32_16x16x32_f16(false, a0, false, bf0.h, (short)0, acc0, false, false);
    acc1 = __builtin_amdgcn_wmma_f32_16x16x32_f16(false, a0, false, bf1.h, (short)0, acc1, false, false);
    acc2 = __builtin_amdgcn_wmma_f32_16x16x32_f16(false, a0, false, bf2.h, (short)0, acc2, false, false);
    acc3 = __builtin_amdgcn_wmma_f32_16x16x32_f16(false, a0, false, bf3.h, (short)0, acc3, false, false);
    acc4 = __builtin_amdgcn_wmma_f32_16x16x32_f16(false, a1, false, bf0.h, (short)0, acc4, false, false);
    acc5 = __builtin_amdgcn_wmma_f32_16x16x32_f16(false, a1, false, bf1.h, (short)0, acc5, false, false);
    acc6 = __builtin_amdgcn_wmma_f32_16x16x32_f16(false, a1, false, bf2.h, (short)0, acc6, false, false);
    acc7 = __builtin_amdgcn_wmma_f32_16x16x32_f16(false, a1, false, bf3.h, (short)0, acc7, false, false);

    if (hasNext) {
#pragma unroll
      for (int e = 0; e < 8; ++e) {
        lBt[nxt][(bc + e) * BT_LD + br0] = f0.h[e];
        lBt[nxt][(bc + e) * BT_LD + br1] = f1.h[e];
      }
      asm volatile("s_wait_asynccnt 0x0" ::: "memory");
    }
    __syncthreads();
    cur = nxt;
  }

  // ---- epilogue
#pragma unroll
  for (int s = 0; s < 2; ++s) {
#pragma unroll
    for (int v = 0; v < 8; ++v) {
      int r = m0 + mt * 32 + s * 16 + v + half * 8;
#pragma unroll
      for (int j = 0; j < 4; ++j) {
        float av;
        if (s == 0) av = (j == 0) ? acc0[v] : (j == 1) ? acc1[v] : (j == 2) ? acc2[v] : acc3[v];
        else        av = (j == 0) ? acc4[v] : (j == 1) ? acc5[v] : (j == 2) ? acc6[v] : acc7[v];
        int c = n0 + nh * 64 + j * 16 + nn;
        float val = av + bias[c];
        if (flags & 2) val += resid[(long)r * N + c];
        if (flags & 1) val = 0.5f * val * (1.f + erff(val * 0.70710678118f));
        if (flags & 4) Cf[(long)r * N + c] = val;
        if (flags & 8) Ch[(long)r * N + c] = (_Float16)val;
      }
    }
  }
}

// ---------------- flash attention (one wave per b,h,16-query tile) --------
// qkv rows: [3E] = Q(0..767) K(768..1535) V(1536..2303), head h at h*64.

__global__ __launch_bounds__(32)
void attn_kernel(const _Float16* __restrict__ qkv, _Float16* __restrict__ o) {
  const int lane = threadIdx.x;
  const int qt = blockIdx.x, h = blockIdx.y, b = blockIdx.z;
  const int nn = lane & 15, half = lane >> 4;

  __shared__ alignas(16) float    S[16][32];
  __shared__ alignas(16) _Float16 Pb[16][32];
  __shared__ alignas(16) _Float16 lV[64 * BT_LD];   // transposed V: [d][key]
  __shared__ float cfac[16];
  __shared__ float linv[16];

  int q_l = qt * 16 + nn; if (q_l > 1024) q_l = 1024;
  const _Float16* qrow = qkv + ((long)(b * NTOK + q_l)) * 2304 + h * 64;
  v16h qa0, qa1;
#pragma unroll
  for (int v = 0; v < 8; ++v) {
    int k = 2 * v + (half << 4);
    qa0[2 * v] = qrow[k];       qa0[2 * v + 1] = qrow[k + 1];
    qa1[2 * v] = qrow[32 + k];  qa1[2 * v + 1] = qrow[32 + k + 1];
  }

  v8f o0 = {}, o1 = {}, o2 = {}, o3 = {};
  float mrun = -1e30f, lrun = 0.f;   // lanes 0..15 own row = lane

  for (int kc = 0; kc < 33; ++kc) {
    const int kbase = kc * 32;

    // ---- stage V tile (32 keys x 64 dims) transposed into LDS
    {
      int key = kbase + lane; int keyc = key > 1024 ? 1024 : key;
      const v4u* vr = (const v4u*)(qkv + ((long)(b * NTOK + keyc)) * 2304 + 1536 + h * 64);
#pragma unroll
      for (int g = 0; g < 8; ++g) {
        F8 f; f.v = vr[g];
#pragma unroll
        for (int e = 0; e < 8; ++e) lV[(g * 8 + e) * BT_LD + lane] = f.h[e];
      }
    }

    // ---- scores for 32 keys (two 16-key WMMA tiles, contraction d=64)
#pragma unroll
    for (int j = 0; j < 2; ++j) {
      int key = kbase + j * 16 + nn;
      int keyc = key > 1024 ? 1024 : key;
      const _Float16* krow = qkv + ((long)(b * NTOK + keyc)) * 2304 + 768 + h * 64;
      v16h kb0, kb1;
#pragma unroll
      for (int v = 0; v < 8; ++v) {   // contiguous 16 f16 runs -> b128 loads
        int d = 2 * v + (half << 4);
        kb0[2 * v] = krow[d];       kb0[2 * v + 1] = krow[d + 1];
        kb1[2 * v] = krow[32 + d];  kb1[2 * v + 1] = krow[32 + d + 1];
      }
      v8f s = {};
      s = __builtin_amdgcn_wmma_f32_16x16x32_f16(false, qa0, false, kb0,
                                                 (short)0, s, false, false);
      s = __builtin_amdgcn_wmma_f32_16x16x32_f16(false, qa1, false, kb1,
                                                 (short)0, s, false, false);
#pragma unroll
      for (int v = 0; v < 8; ++v) {
        int r = v + half * 8;
        int col = j * 16 + nn;
        S[r][col] = (kbase + col <= 1024) ? s[v] * 0.125f : -1e30f;
      }
    }
    __syncthreads();

    // ---- online softmax (lane r < 16 owns row r)
    if (lane < 16) {
      float mx = mrun;
      for (int c = 0; c < 32; ++c) mx = fmaxf(mx, S[lane][c]);
      float corr = __expf(mrun - mx);
      float sum = 0.f;
      for (int c = 0; c < 32; c += 2) {
        float p0 = __expf(S[lane][c] - mx);
        float p1 = __expf(S[lane][c + 1] - mx);
        PK2 pk; pk.h[0] = (_Float16)p0; pk.h[1] = (_Float16)p1;
        *(unsigned*)&Pb[lane][c] = pk.u;
        sum += p0 + p1;
      }
      lrun = lrun * corr + sum;
      mrun = mx;
      cfac[lane] = corr;
    }
    __syncthreads();

    // ---- rescale accumulators by per-row correction
#pragma unroll
    for (int v = 0; v < 8; ++v) {
      float c0 = cfac[v + half * 8];
      o0[v] *= c0; o1[v] *= c0; o2[v] *= c0; o3[v] *= c0;
    }

    // ---- P (16x32) @ V (32x64): hoist all fragments, 4 back-to-back WMMAs
    v16h pa;
#pragma unroll
    for (int v = 0; v < 8; ++v) {
      int k = 2 * v + (half << 4);
      pa[2 * v] = Pb[nn][k]; pa[2 * v + 1] = Pb[nn][k + 1];
    }
    V16 vb0, vb1, vb2, vb3;
    {
      const _Float16* vbp = &lV[nn * BT_LD + 16 * half];
      vb0.u[0] = *(const v4u*)(vbp);              vb0.u[1] = *(const v4u*)(vbp + 8);
      vb1.u[0] = *(const v4u*)(vbp + 16 * BT_LD); vb1.u[1] = *(const v4u*)(vbp + 16 * BT_LD + 8);
      vb2.u[0] = *(const v4u*)(vbp + 32 * BT_LD); vb2.u[1] = *(const v4u*)(vbp + 32 * BT_LD + 8);
      vb3.u[0] = *(const v4u*)(vbp + 48 * BT_LD); vb3.u[1] = *(const v4u*)(vbp + 48 * BT_LD + 8);
    }
    o0 = __builtin_amdgcn_wmma_f32_16x16x32_f16(false, pa, false, vb0.h, (short)0, o0, false, false);
    o1 = __builtin_amdgcn_wmma_f32_16x16x32_f16(false, pa, false, vb1.h, (short)0, o1, false, false);
    o2 = __builtin_amdgcn_wmma_f32_16x16x32_f16(false, pa, false, vb2.h, (short)0, o2, false, false);
    o3 = __builtin_amdgcn_wmma_f32_16x16x32_f16(false, pa, false, vb3.h, (short)0, o3, false, false);
    __syncthreads();   // before next chunk overwrites S/Pb/lV/cfac
  }

  if (lane < 16) linv[lane] = 1.f / lrun;
  __syncthreads();
#pragma unroll
  for (int v = 0; v < 8; ++v) {
    int r = v + half * 8;
    int q = qt * 16 + r;
    if (q <= 1024) {
      float inv = linv[r];
      long base = ((long)(b * NTOK + q)) * 768 + h * 64 + nn;
      o[base +  0] = (_Float16)(o0[v] * inv);
      o[base + 16] = (_Float16)(o1[v] * inv);
      o[base + 32] = (_Float16)(o2[v] * inv);
      o[base + 48] = (_Float16)(o3[v] * inv);
    }
  }
}

// ---------------------------------------------------------------------------

extern "C" void kernel_launch(void* const* d_in, const int* in_sizes, int n_in,
                              void* d_out, int out_size, void* d_ws, size_t ws_size,
                              hipStream_t stream) {
  (void)in_sizes; (void)n_in; (void)out_size; (void)ws_size;
  const float* inp    = (const float*)d_in[0];
  const float* conv_w = (const float*)d_in[1];
  const float* conv_b = (const float*)d_in[2];
  const float* cls_t  = (const float*)d_in[3];
  const float* pos    = (const float*)d_in[4];
  const float* ln1_g  = (const float*)d_in[5];
  const float* ln1_b  = (const float*)d_in[6];
  const float* qkv_w  = (const float*)d_in[7];
  const float* qkv_b  = (const float*)d_in[8];
  const float* proj_w = (const float*)d_in[9];
  const float* proj_b = (const float*)d_in[10];
  const float* ln2_g  = (const float*)d_in[11];
  const float* ln2_b  = (const float*)d_in[12];
  const float* fc1_w  = (const float*)d_in[13];
  const float* fc1_b  = (const float*)d_in[14];
  const float* fc2_w  = (const float*)d_in[15];
  const float* fc2_b  = (const float*)d_in[16];
  const float* fln_g  = (const float*)d_in[17];
  const float* fln_b  = (const float*)d_in[18];
  float* out = (float*)d_out;

  char* base = (char*)d_ws; size_t off = 0;
  auto alloc = [&](size_t bytes) -> void* {
    void* p = base + off;
    off = (off + bytes + 255) & ~(size_t)255;
    return p;
  };
  const long SZ_QKVW = (long)DEPTHc * 768 * 2304;
  const long SZ_PROJ = (long)DEPTHc * 768 * 768;
  const long SZ_FC1  = (long)DEPTHc * 768 * HIDc;
  const long SZ_FC2  = (long)DEPTHc * HIDc * 768;
  _Float16* wq16 = (_Float16*)alloc(SZ_QKVW * 2);
  _Float16* wo16 = (_Float16*)alloc(SZ_PROJ * 2);
  _Float16* w116 = (_Float16*)alloc(SZ_FC1 * 2);
  _Float16* w216 = (_Float16*)alloc(SZ_FC2 * 2);
  _Float16* wc16 = (_Float16*)alloc((long)768 * 768 * 2);
  float*    x    = (float*)   alloc((long)MPAD * 768 * 4);
  _Float16* xh   = (_Float16*)alloc((long)MPAD * 768 * 2);   // LN out / attn out
  _Float16* qkvb = (_Float16*)alloc((long)MPAD * 2304 * 2);
  _Float16* h1   = (_Float16*)alloc((long)MPAD * HIDc * 2);
  _Float16* pm   = (_Float16*)alloc((long)MPAD * 768 * 2);

  auto blks = [](long n) { return (unsigned)((n + 255) / 256); };

  // 1) weight conversion (f32 -> f16)
  cvt_f32_to_f16<<<blks(SZ_QKVW), 256, 0, stream>>>(qkv_w, wq16, SZ_QKVW);
  cvt_f32_to_f16<<<blks(SZ_PROJ), 256, 0, stream>>>(proj_w, wo16, SZ_PROJ);
  cvt_f32_to_f16<<<blks(SZ_FC1),  256, 0, stream>>>(fc1_w,  w116, SZ_FC1);
  cvt_f32_to_f16<<<blks(SZ_FC2),  256, 0, stream>>>(fc2_w,  w216, SZ_FC2);
  cvt_transpose_768<<<blks((long)768 * 768), 256, 0, stream>>>(conv_w, wc16);

  // 2) patch embed: im2col + GEMM (+bias) -> x ; then cls/pos fix
  im2col_kernel<<<blks((long)MPAD * 768), 256, 0, stream>>>(inp, pm);
  gemm_ws<<<dim3(768 / 128, MPAD / 128), 256, 0, stream>>>(
      pm, wc16, conv_b, nullptr, x, nullptr, 768, 768, /*f32*/4);
  embed_fix_kernel<<<blks((long)MPAD * 768), 256, 0, stream>>>(x, cls_t, pos);

  // 3) transformer blocks
  for (int l = 0; l < DEPTHc; ++l) {
    ln_kernel<<<MPAD, 256, 0, stream>>>(x, ln1_g + (long)l * 768,
                                        ln1_b + (long)l * 768, xh);
    gemm_ws<<<dim3(2304 / 128, MPAD / 128), 256, 0, stream>>>(
        xh, wq16 + (long)l * 768 * 2304, qkv_b + (long)l * 2304,
        nullptr, nullptr, qkvb, 2304, 768, /*f16*/8);
    attn_kernel<<<dim3(65, NHc, 4), 32, 0, stream>>>(qkvb, xh);
    gemm_ws<<<dim3(768 / 128, MPAD / 128), 256, 0, stream>>>(
        xh, wo16 + (long)l * 768 * 768, proj_b + (long)l * 768,
        x, x, nullptr, 768, 768, /*resid|f32*/2 | 4);
    ln_kernel<<<MPAD, 256, 0, stream>>>(x, ln2_g + (long)l * 768,
                                        ln2_b + (long)l * 768, xh);
    gemm_ws<<<dim3(HIDc / 128, MPAD / 128), 256, 0, stream>>>(
        xh, w116 + (long)l * 768 * HIDc, fc1_b + (long)l * HIDc,
        nullptr, nullptr, h1, HIDc, 768, /*gelu|f16*/1 | 8);
    gemm_ws<<<dim3(768 / 128, MPAD / 128), 256, 0, stream>>>(
        h1, w216 + (long)l * HIDc * 768, fc2_b + (long)l * 768,
        x, x, nullptr, 768, HIDc, /*resid|f32*/2 | 4);
  }

  // 4) final LN + NCHW transpose store
  final_kernel<<<4096, 256, 0, stream>>>(x, fln_g, fln_b, out);
}
